// Autoregression_32650341384227
// MI455X (gfx1250) — matrix operand order repaired and checked by reading, not compiled, not run on hardware
//
#include <hip/hip_runtime.h>
#include <hip/hip_bf16.h>
#include <math.h>

typedef __attribute__((ext_vector_type(16))) _Float16 v16h;
typedef __attribute__((ext_vector_type(8)))  _Float16 v8h;
typedef __attribute__((ext_vector_type(8)))  float    v8f;

#define C_DIM 64
#define T_DIM 65536
#define K_ST  16
#define TAPS  9
#define KDIM  576          // 9 taps * 64 cin
#define NQ    18           // 576 / 32
#define NCT   4            // 64 couts / 16

#define XT_ROWS (T_DIM + 8)

// workspace layout (bytes)
#define OFF_XT    0ull
#define OFF_M     (OFF_XT + (size_t)XT_ROWS * C_DIM * 2)              // f32 [K][64][64]
#define OFF_U     (OFF_M  + (size_t)K_ST * C_DIM * C_DIM * 4)         // f32 [K][64][576]
#define OFF_BW    (OFF_U  + (size_t)K_ST * C_DIM * KDIM * 4)          // f16 packed W'
#define OFF_BU    (OFF_BW + (size_t)K_ST * NCT * NQ * 32 * 16 * 2)    // f16 packed U
#define OFF_MB    (OFF_BU + (size_t)K_ST * NCT * NQ * 32 * 16 * 2)    // f32 [K][64]
#define OFF_CK    (OFF_MB + (size_t)K_ST * C_DIM * 4)                 // f32 [K]

// ---------- Phase 0: transpose + f16-convert x with 8 rows of causal zero pad ----------
__global__ void k_xT(const float* __restrict__ x, _Float16* __restrict__ xT) {
  int idx = blockIdx.x * blockDim.x + threadIdx.x;
  if (idx >= XT_ROWS * C_DIM) return;
  int r = idx >> 6, c = idx & 63;
  float v = (r < 8) ? 0.f : x[(size_t)c * T_DIM + (r - 8)];
  xT[idx] = (_Float16)v;
}

// ---------- Phase 1a: per-k Cholesky, logdet, Sigma^-1 (tiny: 16 x 64x64) ----------
__global__ void k_cholinv(const float* __restrict__ Sigma, float* __restrict__ Minv,
                          float* __restrict__ constk) {
  __shared__ float S[64][65];
  __shared__ float Y[64][65];
  int k = blockIdx.x, tid = threadIdx.x;
  for (int i = 0; i < 64; ++i) S[i][tid] = Sigma[(size_t)k * 4096 + i * 64 + tid];
  __syncthreads();
  for (int j = 0; j < 64; ++j) {
    if (tid == j) {
      float d = S[j][j];
      for (int p = 0; p < j; ++p) d -= S[j][p] * S[j][p];
      S[j][j] = sqrtf(d);
    }
    __syncthreads();
    if (tid > j) {
      float s = S[tid][j];
      for (int p = 0; p < j; ++p) s -= S[tid][p] * S[j][p];
      S[tid][j] = s / S[j][j];
    }
    __syncthreads();
  }
  if (tid == 0) {
    float ld = 0.f;
    for (int i = 0; i < 64; ++i) ld += logf(S[i][i]);
    constk[k] = -0.5f * (64.f * logf(6.283185307179586f) + 2.f * ld);
  }
  // thread tid solves column tid of Sigma^-1: forward L y = e_tid, then L^T z = y
  for (int i = 0; i < 64; ++i) {
    float s = (i == tid) ? 1.f : 0.f;
    for (int p = 0; p < i; ++p) s -= S[i][p] * Y[p][tid];
    Y[i][tid] = s / S[i][i];
  }
  for (int i = 63; i >= 0; --i) {
    float s = Y[i][tid];
    for (int p = i + 1; p < 64; ++p) s -= S[p][i] * Y[p][tid];
    Y[i][tid] = s / S[i][i];
  }
  for (int i = 0; i < 64; ++i) Minv[(size_t)k * 4096 + i * 64 + tid] = Y[i][tid];
}

// ---------- Phase 1b: U[k] = Sigma^-1 * W'  (f32, tiny) ----------
__global__ void k_U(const float* __restrict__ Minv, const float* __restrict__ W,
                    float* __restrict__ U) {
  int idx = blockIdx.x * blockDim.x + threadIdx.x;
  if (idx >= K_ST * C_DIM * KDIM) return;
  int jc = idx % KDIM;
  int r  = (idx / KDIM) % C_DIM;
  int k  = idx / (KDIM * C_DIM);
  int j = jc >> 6, cin = jc & 63;
  const float* Mrow = Minv + ((size_t)k * C_DIM + r) * C_DIM;
  const float* Wp = W + (size_t)k * C_DIM * C_DIM * TAPS + (size_t)cin * TAPS + j;
  float s = 0.f;
  for (int c = 0; c < C_DIM; ++c) s += Mrow[c] * Wp[(size_t)c * C_DIM * TAPS];
  U[idx] = s;
}

// ---------- Phase 1c: prepack B operands into exact wave32 WMMA lane layout ----------
// B[kk, n] for chunk q: lane = n + 16*((kk/8)%2), element i = kk%8 + 8*(kk/16)
__global__ void k_pack(const float* __restrict__ W, const float* __restrict__ U,
                       _Float16* __restrict__ BW, _Float16* __restrict__ BU) {
  int idx = blockIdx.x * blockDim.x + threadIdx.x;
  if (idx >= K_ST * NCT * NQ * 32 * 16) return;
  int i    = idx & 15;
  int lane = (idx >> 4) & 31;
  int rest = idx >> 9;
  int q  = rest % NQ;
  int r2 = rest / NQ;
  int ct = r2 & 3;
  int k  = r2 >> 2;
  int cout = ct * 16 + (lane & 15);
  int kg = q * 32 + ((i >> 3) << 4) + ((lane >> 4) << 3) + (i & 7);   // global im2col K index (jc = j*64+cin)
  int j = kg >> 6, cin = kg & 63;
  BW[idx] = (_Float16)W[(((size_t)k * C_DIM + cout) * C_DIM + cin) * TAPS + j];
  BU[idx] = (_Float16)U[((size_t)k * C_DIM + cout) * KDIM + kg];
}

// ---------- Phase 1d: Mb[k] = Sigma^-1 b[k] ----------
__global__ void k_mb(const float* __restrict__ Minv, const float* __restrict__ bvec,
                     float* __restrict__ Mb) {
  int idx = blockIdx.x * blockDim.x + threadIdx.x;
  if (idx >= K_ST * C_DIM) return;
  int c = idx & 63, k = idx >> 6;
  const float* Mrow = Minv + ((size_t)k * C_DIM + c) * C_DIM;
  const float* bk = bvec + k * C_DIM;
  float s = 0.f;
  for (int p = 0; p < C_DIM; ++p) s += Mrow[p] * bk[p];
  Mb[idx] = s;
}

#define WMMA_F16(A, B, Cacc) \
  __builtin_amdgcn_wmma_f32_16x16x32_f16(false, (A), false, (B), (short)0, (Cacc), false, false)

// ---------- Phase 2: fused conv + quadratic form, all WMMA ----------
// wave = one (k, 32-wide t block = two 16-row strips sharing B tiles).
// acc_e = E^T tiles, acc_g = G^T tiles (G = Sigma^-1 E).
__global__ __launch_bounds__(256)
void k_main(const _Float16* __restrict__ xT,
            const _Float16* __restrict__ BW,
            const _Float16* __restrict__ BU,
            const float* __restrict__ bvec,
            const float* __restrict__ Mb,
            const float* __restrict__ constk,
            float* __restrict__ out) {
  int lane = threadIdx.x & 31;
  int wid  = blockIdx.x * (blockDim.x >> 5) + (threadIdx.x >> 5);
  int k  = wid & (K_ST - 1);          // 8 waves of a block share the t range across k
  int tb = (wid >> 4) << 5;           // 32 t's per wave (two strips)

  int m  = lane & 15;                 // A-matrix row (t offset)
  int hf = lane >> 4;
  int c0base = hf << 3;               // 16-bit A layout: K = 8*hf + i (i<8), i+8+8*hf (i>=8)

  v8f zero = {};
  v8f acc_e[2][NCT], acc_g[2][NCT];
#pragma unroll
  for (int u = 0; u < 2; ++u)
#pragma unroll
    for (int ct = 0; ct < NCT; ++ct) { acc_e[u][ct] = zero; acc_g[u][ct] = zero; }

  const _Float16* BWk = BW + (size_t)k * NCT * NQ * 512;
  const _Float16* BUk = BU + (size_t)k * NCT * NQ * 512;

  for (int q = 0; q < NQ; ++q) {
    int j  = q >> 1;
    int c0 = ((q & 1) << 5) + c0base;
    const _Float16* ap0 = xT + ((size_t)(tb + m + j) << 6) + c0;
    const _Float16* ap1 = ap0 + (16 << 6);          // second 16-row strip
    v8h a0lo = *(const v8h*)ap0;
    v8h a0hi = *(const v8h*)(ap0 + 16);
    v8h a1lo = *(const v8h*)ap1;
    v8h a1hi = *(const v8h*)(ap1 + 16);
    v16h A0 = __builtin_shufflevector(a0lo, a0hi, 0,1,2,3,4,5,6,7,8,9,10,11,12,13,14,15);
    v16h A1 = __builtin_shufflevector(a1lo, a1hi, 0,1,2,3,4,5,6,7,8,9,10,11,12,13,14,15);

    // ---- W' matmul: one load clause of 4 independent B tiles, then 8 wmma ----
    v16h Bw[NCT];
#pragma unroll
    for (int ct = 0; ct < NCT; ++ct)
      Bw[ct] = *(const v16h*)(BWk + (((ct * NQ + q) * 32 + lane) << 4));
#pragma unroll
    for (int ct = 0; ct < NCT; ++ct) {
      acc_e[0][ct] = WMMA_F16(A0, Bw[ct], acc_e[0][ct]);
      acc_e[1][ct] = WMMA_F16(A1, Bw[ct], acc_e[1][ct]);
    }

    // ---- U = Sigma^-1 W' matmul: same A, second set of B tiles ----
    v16h Bu[NCT];
#pragma unroll
    for (int ct = 0; ct < NCT; ++ct)
      Bu[ct] = *(const v16h*)(BUk + (((ct * NQ + q) * 32 + lane) << 4));
#pragma unroll
    for (int ct = 0; ct < NCT; ++ct) {
      acc_g[0][ct] = WMMA_F16(A0, Bu[ct], acc_g[0][ct]);
      acc_g[1][ct] = WMMA_F16(A1, Bu[ct], acc_g[1][ct]);
    }
  }

  // C layout: VGPR p, lanes0-15 -> (t = base+p, n = lane), lanes16-31 -> (t = base+p+8, n = lane-16)
  int n = lane & 15;
  float ck = constk[k];
#pragma unroll
  for (int u = 0; u < 2; ++u) {
    float s[8];
#pragma unroll
    for (int p = 0; p < 8; ++p) s[p] = 0.f;
#pragma unroll
    for (int ct = 0; ct < NCT; ++ct) {
      float be = bvec[k * C_DIM + ct * 16 + n];
      float bg = Mb[k * C_DIM + ct * 16 + n];
#pragma unroll
      for (int p = 0; p < 8; ++p) {
        float e = acc_e[u][ct][p] + be;
        float g = acc_g[u][ct][p] + bg;
        s[p] += e * g;
      }
    }
    // reduce over the 16 cout lanes within each half-wave
#pragma unroll
    for (int p = 0; p < 8; ++p) {
      s[p] += __shfl_xor(s[p], 1, 32);
      s[p] += __shfl_xor(s[p], 2, 32);
      s[p] += __shfl_xor(s[p], 4, 32);
      s[p] += __shfl_xor(s[p], 8, 32);
    }
    if (n == 0) {
      int t0 = tb + (u << 4) + (hf << 3);
#pragma unroll
      for (int p = 0; p < 8; ++p)
        out[(size_t)k * T_DIM + t0 + p] = ck - 0.5f * s[p];
    }
  }
}

extern "C" void kernel_launch(void* const* d_in, const int* in_sizes, int n_in,
                              void* d_out, int out_size, void* d_ws, size_t ws_size,
                              hipStream_t stream) {
  const float* x     = (const float*)d_in[0];
  const float* W     = (const float*)d_in[1];
  const float* b     = (const float*)d_in[2];
  const float* Sigma = (const float*)d_in[3];
  float* out = (float*)d_out;
  char* ws = (char*)d_ws;

  _Float16* xT = (_Float16*)(ws + OFF_XT);
  float* Minv  = (float*)(ws + OFF_M);
  float* U     = (float*)(ws + OFF_U);
  _Float16* BW = (_Float16*)(ws + OFF_BW);
  _Float16* BU = (_Float16*)(ws + OFF_BU);
  float* Mb    = (float*)(ws + OFF_MB);
  float* ck    = (float*)(ws + OFF_CK);

  {
    int nel = XT_ROWS * C_DIM;
    k_xT<<<(nel + 255) / 256, 256, 0, stream>>>(x, xT);
  }
  k_cholinv<<<K_ST, 64, 0, stream>>>(Sigma, Minv, ck);
  {
    int nel = K_ST * C_DIM * KDIM;
    k_U<<<(nel + 255) / 256, 256, 0, stream>>>(Minv, W, U);
  }
  {
    int nel = K_ST * NCT * NQ * 32 * 16;
    k_pack<<<(nel + 255) / 256, 256, 0, stream>>>(W, U, BW, BU);
  }
  {
    int nel = K_ST * C_DIM;
    k_mb<<<(nel + 255) / 256, 256, 0, stream>>>(Minv, b, Mb);
  }
  {
    int waves = K_ST * (T_DIM / 32);      // 32768 wave-tiles (32 t's each)
    int blocks = waves / 8;               // 8 waves / block
    k_main<<<blocks, 256, 0, stream>>>(xT, BW, BU, b, Mb, ck, out);
  }
}
